// NetVLAD_16707422781749
// MI455X (gfx1250) — matrix operand order repaired and checked by reading, not compiled, not run on hardware
//
#include <hip/hip_runtime.h>
#include <hip/hip_bf16.h>
#include <math.h>
#include <stdint.h>

// NetVLAD for MI455X (gfx1250, wave32).
//   x:        [8,128,32,32] f32
//   conv_w:   [68,128]      f32
//   centroids:[68,128]      f32
// out = concat( store_residual [8,64,128,32,32], soft_assign [8,68,32,32] )

#define C_DIM 128
#define HW    1024          // 32*32
#define NBAT  8
#define KTOT  68
#define KREAL 64
#define KPAD  80            // 68 padded to 5 WMMA tiles of 16
#define PIX_TOT 8192        // NBAT*HW
#define EPSN  1e-12f

typedef __attribute__((ext_vector_type(16))) _Float16 v16h;
typedef __attribute__((ext_vector_type(8)))  float    v8f;

// ---------------------------------------------------------------------------
// K0: per-pixel inverse L2 norm over channels.  32 blocks x 256 threads.
// Threads walk consecutive hw -> fully coalesced b32 loads for each channel.
// ---------------------------------------------------------------------------
__global__ void nv_rnorm(const float* __restrict__ x, float* __restrict__ rnorm)
{
    int p  = blockIdx.x * 256 + threadIdx.x;        // global pixel 0..8191
    int n  = p >> 10;
    int hw = p & 1023;
    const float* xp = x + (size_t)n * C_DIM * HW + hw;
    float s = 0.0f;
#pragma unroll 8
    for (int c = 0; c < C_DIM; ++c) {
        float v = xp[(size_t)c * HW];
        s += v * v;
    }
    rnorm[p] = 1.0f / fmaxf(sqrtf(s), EPSN);
}

// ---------------------------------------------------------------------------
// K1: soft-assignment.  64 blocks x 256 threads (8 waves); block = 128 pixels.
// LDS phase 1: xh  f16 [128c][128p] (32KB) + cwh f16 [80k][128c] (20KB)
// LDS phase 2 (aliased after barrier): lg f32 [128p][80k] (40KB)
// Each wave: 16 pixels x 80 clusters via 5x4 = 20 v_wmma_f32_16x16x32_f16.
// ---------------------------------------------------------------------------
__global__ void nv_assign(const float* __restrict__ x,
                          const float* __restrict__ conv_w,
                          const float* __restrict__ rnorm,
                          float* __restrict__ sa_out)
{
    extern __shared__ char smem[];
    _Float16* xh  = (_Float16*)smem;             // 32768 B
    _Float16* cwh = (_Float16*)(smem + 32768);   // 20480 B
    float*    lg  = (float*)smem;                // 40960 B, aliased later

    const int t     = threadIdx.x;
    const int b     = blockIdx.x;                // 0..63
    const int n     = b >> 3;
    const int hw0   = (b & 7) * 128;
    const int pbase = b * 128;

    // --- stage normalized x as f16: xh[c*128 + p] ---
    {
        int p  = t & 127;
        int c0 = t >> 7;                         // 0 or 1
        float r = rnorm[pbase + p];
        const float* xp = x + (size_t)n * C_DIM * HW + hw0 + p;
        for (int c = c0; c < C_DIM; c += 2)
            xh[c * 128 + p] = (_Float16)(xp[(size_t)c * HW] * r);
    }
    // --- stage conv_w as f16, zero-padded to 80 cluster rows ---
    for (int idx = t; idx < KPAD * C_DIM; idx += 256) {
        int k = idx >> 7, c = idx & 127;
        cwh[idx] = (_Float16)((k < KTOT) ? conv_w[k * C_DIM + c] : 0.0f);
    }
    __syncthreads();

    // --- WMMA: wave wv owns pixels [wv*16, wv*16+16) ---
    const int lane = t & 31;
    const int wv   = t >> 5;
    const int m    = lane & 15;
    const int pl   = wv * 16 + m;                // local pixel of this lane
    const int koff = (lane < 16) ? 0 : 8;        // A-matrix 16x32 f16 layout

    v16h A[4];
#pragma unroll
    for (int kb = 0; kb < 4; ++kb) {
        int cb = kb * 32;
#pragma unroll
        for (int e = 0; e < 8; ++e) A[kb][e]     = xh[(cb + koff + e) * 128 + pl];
#pragma unroll
        for (int e = 0; e < 8; ++e) A[kb][8 + e] = xh[(cb + 16 + koff + e) * 128 + pl];
    }

    const int nn   = lane & 15;                  // B/D column
    const int kb2  = (lane < 16) ? 0 : 8;        // B 32x16: lanes16-31 hold K=8..15,24..31
    v8f accs[5];
#pragma unroll
    for (int nt = 0; nt < 5; ++nt) {
        v8f acc = {};
        int nb = nt * 16;
#pragma unroll
        for (int kb = 0; kb < 4; ++kb) {
            v16h B;
            const _Float16* bp = cwh + (nb + nn) * C_DIM + kb * 32 + kb2;
#pragma unroll
            for (int e = 0; e < 16; ++e) B[e] = bp[(e < 8) ? e : 8 + e];
            acc = __builtin_amdgcn_wmma_f32_16x16x32_f16(
                      false, A[kb], false, B, (short)0, acc, false, false);
        }
        accs[nt] = acc;
    }
    __syncthreads();                             // all waves done with xh/cwh

    // --- dump logits into aliased LDS region ---
    const int mrow = (lane < 16) ? 0 : 8;        // D layout: VGPR v -> row v(+8)
#pragma unroll
    for (int nt = 0; nt < 5; ++nt)
#pragma unroll
        for (int v = 0; v < 8; ++v)
            lg[(wv * 16 + mrow + v) * KPAD + nt * 16 + nn] = accs[nt][v];
    __syncthreads();

    // --- softmax over 68 clusters; one thread per pixel, coalesced writes ---
    if (t < 128) {
        float* row = lg + t * KPAD;
        float mx = row[0];
        for (int k = 1; k < KTOT; ++k) mx = fmaxf(mx, row[k]);
        float s = 0.0f;
        for (int k = 0; k < KTOT; ++k) { float e = expf(row[k] - mx); row[k] = e; s += e; }
        float inv = 1.0f / s;
        float* o = sa_out + (size_t)n * KTOT * HW + hw0 + t;
        for (int k = 0; k < KTOT; ++k) o[(size_t)k * HW] = row[k] * inv;
    }
}

// ---------------------------------------------------------------------------
// K2: fused residual, the 268 MB streaming write (bandwidth bound).
// Grid (64 hw-blocks, 8 k-blocks) x 256 threads.  Raw x tile (128c x 128hw,
// 64KB) staged straight into LDS with GLOBAL_LOAD_ASYNC_TO_LDS_B128 (no VGPR
// round trip, tracked by ASYNCcnt); rnorm is a per-thread constant applied at
// use.  Output goes out through non-temporal b32 stores (write-once data).
// ---------------------------------------------------------------------------
__global__ void nv_resid(const float* __restrict__ x,
                         const float* __restrict__ cent,
                         const float* __restrict__ rnorm,
                         const float* __restrict__ sa,
                         float* __restrict__ out)
{
    extern __shared__ float xs[];                // [128c][128p], 65536 B
    const int t     = threadIdx.x;
    const int hwb   = blockIdx.x;                // 0..63
    const int kb    = blockIdx.y;                // 0..7
    const int n     = hwb >> 3;
    const int hw0   = (hwb & 7) * 128;
    const int pbase = hwb * 128;
    const int p  = t & 127;
    const int c0 = t >> 7;                       // 0 or 1

    // --- async DMA of the raw x tile into LDS: 4096 x b128 transfers ---
    {
        const float* gbase = x + (size_t)n * C_DIM * HW + hw0;
        uint32_t lds0 = (uint32_t)(uintptr_t)(void*)xs;  // LDS byte offset
#pragma unroll
        for (int j = 0; j < 16; ++j) {
            int i   = j * 256 + t;               // 0..4095
            int c   = i >> 5;                    // channel row (512 B each)
            int seg = i & 31;                    // 16-byte segment within row
            uint32_t ldsa = lds0 + (uint32_t)(c * 512 + seg * 16);
            uint32_t goff = (uint32_t)(c * 4096 + seg * 16);
            asm volatile("global_load_async_to_lds_b128 %0, %1, %2"
                         :
                         : "v"(ldsa), "v"(goff), "s"(gbase)
                         : "memory");
        }
#if __has_builtin(__builtin_amdgcn_s_wait_asynccnt)
        __builtin_amdgcn_s_wait_asynccnt(0);
#else
        asm volatile("s_wait_asynccnt 0" ::: "memory");
#endif
    }
    __syncthreads();

    const float r = rnorm[pbase + p];            // per-thread pixel constant
    const float* sab = sa + (size_t)n * KTOT * HW + hw0 + p;
    for (int kk = 0; kk < 8; ++kk) {
        int k = kb * 8 + kk;
        float a = sab[(size_t)k * HW];           // soft assignment, coalesced
        const float* ck = cent + k * C_DIM;      // broadcast loads (L1-hot)
        float* ob = out + (((size_t)(n * KREAL + k)) * C_DIM + c0) * HW + hw0 + p;
#pragma unroll 4
        for (int c2 = 0; c2 < 64; ++c2) {
            int c = c0 + c2 * 2;
            float v = (xs[c * 128 + p] * r - ck[c]) * a;
            __builtin_nontemporal_store(v, ob + (size_t)c2 * 2 * HW);
        }
    }
}

// ---------------------------------------------------------------------------
extern "C" void kernel_launch(void* const* d_in, const int* in_sizes, int n_in,
                              void* d_out, int out_size, void* d_ws, size_t ws_size,
                              hipStream_t stream)
{
    const float* x      = (const float*)d_in[0];
    const float* conv_w = (const float*)d_in[1];
    const float* cent   = (const float*)d_in[2];

    float* out   = (float*)d_out;
    float* sa    = out + (size_t)NBAT * KREAL * C_DIM * HW;  // +67,108,864
    float* rnorm = (float*)d_ws;                             // 32 KB scratch

    nv_rnorm<<<PIX_TOT / 256, 256, 0, stream>>>(x, rnorm);

    // dyn LDS: max(32KB xh + 20KB cwh, 40KB logits) with aliasing = 53248 B
    nv_assign<<<64, 256, 53248, stream>>>(x, conv_w, rnorm, sa);

    dim3 g2(64, 8);
    nv_resid<<<g2, 256, 65536, stream>>>(x, cent, rnorm, sa, out);
}
